// RefNet_29892972380447
// MI455X (gfx1250) — matrix operand order repaired and checked by reading, not compiled, not run on hardware
//
#include <hip/hip_runtime.h>
#include <hip/hip_bf16.h>

// Sizes from the reference
#define BB        4
#define N_POINTS  524288
#define NUM_PROP  256
#define FEAT_DIM  16
#define K_INST    32
#define P_TOT     256

// Output layout (floats), concatenated in reference return order:
//   batch        : [0, 16384)        (4*256*16)
//   proposal_mask: [16384, 17408)    (4*256)
//   scores_pred  : [17408, 17664)    (256)
//   proposal_ious: [17664, 50432)    (4*256*32)
//   proposals_pred:[50432, 50432 + 256*524288)
#define OFF_BATCH 0
#define OFF_MASK  16384
#define OFF_SPRED 17408
#define OFF_IOUS  17664
#define OFF_PRED  50432

typedef __attribute__((ext_vector_type(16))) _Float16 v16h;
typedef __attribute__((ext_vector_type(8)))  float    v8f;

// ---------------------------------------------------------------------------
// Kernel 1: zero the entire output (512+ MB). Pure HBM-write bound; float4
// stores lower to global_store_b128. This is the dominant cost (~23 us floor
// at 23.3 TB/s) — nothing smarter to do than saturate store bandwidth.
// ---------------------------------------------------------------------------
__global__ void zero_fill_kernel(float4* __restrict__ out, long n4) {
    long i = (long)blockIdx.x * blockDim.x + threadIdx.x;
    if (i < n4) out[i] = make_float4(0.f, 0.f, 0.f, 0.f);
}

// ---------------------------------------------------------------------------
// Kernel 2: scatter 1.0 into proposals_pred at (pid, ptid). 1M random 4B
// stores — inherently scatter-bound; int2 load keeps the index read coalesced.
// ---------------------------------------------------------------------------
__global__ void scatter_ones_kernel(const int* __restrict__ pidx,
                                    float* __restrict__ pred, int M) {
    int i = blockIdx.x * blockDim.x + threadIdx.x;
    if (i < M) {
        int2 row = ((const int2*)pidx)[i];          // (pid, ptid)
        pred[(long)row.x * N_POINTS + row.y] = 1.0f;
    }
}

// ---------------------------------------------------------------------------
// Kernel 3: per-proposal metadata. One workgroup (256 threads = 8 wave32).
//   Phase 1: b_idx[p] via searchsorted over batch_offsets (5 entries).
//   Phase 2: slot[p] via cumsum(onehot) computed as a lower-triangular matmul
//            D(256x16) = L(256x256) @ onehot(256x16pad) using
//            v_wmma_f32_16x16x32_f16. Wave w owns row-tiles 2w and 2w+1;
//            tile t accumulates over K-chunks c = 0 .. t/2 (32 rows each).
//   Phase 3: scatter feats/mask/ious by (b_idx, slot); sigmoid(scores).
// ---------------------------------------------------------------------------
__global__ __launch_bounds__(256) void proposal_meta_kernel(
    const float* __restrict__ scores,
    const float* __restrict__ feats,
    const float* __restrict__ ious,
    const int*   __restrict__ pidx,
    const int*   __restrict__ poff,
    const int*   __restrict__ boff,
    float* __restrict__ out)
{
    __shared__ int lds_b[P_TOT];
    __shared__ int lds_slot[P_TOT];

    const int p = threadIdx.x;

    // --- Phase 1: batch index of each proposal -----------------------------
    {
        int first = pidx[2 * poff[p] + 1];          // first point id of proposal p
        int b = 0;
        #pragma unroll
        for (int j = 1; j < BB; ++j) if (first >= boff[j]) b = j;
        lds_b[p] = b;
    }
    __syncthreads();

    // --- Phase 2: inclusive cumsum of one-hot via triangular WMMA ----------
    const int lane   = threadIdx.x & 31;
    const int laneHi = lane >> 4;                   // 0: lanes 0-15, 1: lanes 16-31
    const int wave   = threadIdx.x >> 5;

    for (int t = wave * 2; t < wave * 2 + 2; ++t) {
        v8f acc = {};
        const int mrow   = (lane & 15) + t * 16;    // global M row for A fragment
        const int ncol   = lane & 15;               // global N col for B fragment
        const int chunks = (t >> 1) + 1;            // K-chunks of 32 covering rows <= 16t+15
        for (int c = 0; c < chunks; ++c) {
            // A fragment: 16x32 f16, A[m,k] = (k <= m)  (inclusive lower-triangular L)
            // layout: elem e -> K = 16*(e>=8) + 8*laneHi + 2*((e>>1)&3) + (e&1)
            v16h a;
            #pragma unroll
            for (int e = 0; e < 16; ++e) {
                int k_in = ((e >> 3) << 4) + (laneHi << 3) + (((e >> 1) & 3) << 1) + (e & 1);
                int kg   = c * 32 + k_in;
                a[e] = (kg <= mrow) ? (_Float16)1.0f : (_Float16)0.0f;
            }
            // B fragment: 32x16 f16, B[k,n] = onehot(b_idx[k]) padded to 16 cols
            // layout: elem e -> K = 16*(e>=8) + 8*laneHi + (e&7), N = lane&15
            v16h bm;
            #pragma unroll
            for (int e = 0; e < 16; ++e) {
                int kg = c * 32 + ((e >> 3) << 4) + (laneHi << 3) + (e & 7);
                bm[e] = (lds_b[kg] == ncol) ? (_Float16)1.0f : (_Float16)0.0f;
            }
            acc = __builtin_amdgcn_wmma_f32_16x16x32_f16(
                false, a, false, bm, (short)0, acc, false, false);
        }
        // Extract: C/D layout lane l -> N = l&15, vgpr r -> M = r + 8*laneHi.
        // slot[p] = cumsum[p, b_idx[p]] - 1  (exact small ints in f32).
        #pragma unroll
        for (int r = 0; r < 8; ++r) {
            int pp = t * 16 + r + (laneHi << 3);
            if ((lane & 15) == lds_b[pp])
                lds_slot[pp] = (int)(acc[r] + 0.5f) - 1;
        }
    }
    __syncthreads();

    // --- Phase 3: scatter per-proposal data --------------------------------
    {
        const int b   = lds_b[p];
        const int sl  = lds_slot[p];
        const int row = b * NUM_PROP + sl;

        out[OFF_SPRED + p] = 1.0f / (1.0f + __expf(-scores[p]));
        out[OFF_MASK + row] = 1.0f;
        #pragma unroll
        for (int f = 0; f < FEAT_DIM; ++f)
            out[OFF_BATCH + row * FEAT_DIM + f] = feats[p * FEAT_DIM + f];
        #pragma unroll
        for (int k = 0; k < K_INST; ++k)
            out[OFF_IOUS + row * K_INST + k] = ious[p * K_INST + k];
    }
}

// ---------------------------------------------------------------------------
extern "C" void kernel_launch(void* const* d_in, const int* in_sizes, int n_in,
                              void* d_out, int out_size, void* d_ws, size_t ws_size,
                              hipStream_t stream) {
    const float* scores = (const float*)d_in[0];
    const float* feats  = (const float*)d_in[1];
    const float* ious   = (const float*)d_in[2];
    const int*   pidx   = (const int*)d_in[3];   // (M,2) int32: (pid, ptid)
    const int*   poff   = (const int*)d_in[4];   // (P+1,)
    const int*   boff   = (const int*)d_in[5];   // (B+1,)
    float* out = (float*)d_out;

    // 1) Zero the whole output (covers batch/mask/ious/pred; spred overwritten).
    long n4 = (long)out_size / 4;                // out_size is a multiple of 4
    long zblocks = (n4 + 255) / 256;
    zero_fill_kernel<<<(unsigned)zblocks, 256, 0, stream>>>((float4*)out, n4);

    // 2) Scatter ones into proposals_pred (stream order guarantees zero first).
    int M = in_sizes[3] / 2;
    scatter_ones_kernel<<<(M + 255) / 256, 256, 0, stream>>>(pidx, out + OFF_PRED, M);

    // 3) Per-proposal metadata: WMMA prefix-sum + scatters (single WGP).
    proposal_meta_kernel<<<1, 256, 0, stream>>>(scores, feats, ious,
                                                pidx, poff, boff, out);
}
